// TransformerBlock_1803886265632
// MI455X (gfx1250) — matrix operand order repaired and checked by reading, not compiled, not run on hardware
//
#include <hip/hip_runtime.h>

typedef __attribute__((ext_vector_type(16))) __bf16 v16bf;
typedef __attribute__((ext_vector_type(8)))  __bf16 v8bf;
typedef __attribute__((ext_vector_type(8)))  float  v8f;
typedef int      v4i_vec __attribute__((vector_size(16)));
typedef unsigned v4u_vec __attribute__((vector_size(16)));
typedef int      v8i_vec __attribute__((vector_size(32)));
typedef __attribute__((address_space(1))) v4i_vec* gbl_v4i_ptr;
typedef __attribute__((address_space(3))) v4i_vec* lds_v4i_ptr;

#define WMMA_BF16(a, b, c) \
  __builtin_amdgcn_wmma_f32_16x16x32_bf16(false, (a), false, (b), (short)0, (c), false, false)

// ---- gfx1250 async global->LDS + TDM (signatures verified by compile probes) ----
#if defined(__has_builtin)
#if __has_builtin(__builtin_amdgcn_global_load_async_to_lds_b128)
#define HAVE_ASYNC_LDS 1
#endif
#if __has_builtin(__builtin_amdgcn_tensor_load_to_lds) && \
    __has_builtin(__builtin_amdgcn_s_wait_tensorcnt)
#define HAVE_TDM 1
#endif
#endif

#ifdef HAVE_ASYNC_LDS
#if __has_builtin(__builtin_amdgcn_s_wait_asynccnt)
#define WAIT_ASYNC(n) __builtin_amdgcn_s_wait_asynccnt(n)
#else
#define WAIT_ASYNC(n) asm volatile("s_wait_asynccnt %0" ::"i"(n) : "memory")
#endif
static __device__ __forceinline__ void async_cp128(const __bf16* g, __bf16* l) {
  __builtin_amdgcn_global_load_async_to_lds_b128(
      (gbl_v4i_ptr)(void*)g, (lds_v4i_ptr)(void*)l, 0, 0);
}
#endif

static __device__ __forceinline__ unsigned lds_addr_of(void* p) {
  return (unsigned)(size_t)(__attribute__((address_space(3))) void*)p;
}

#ifdef HAVE_TDM
#define WAIT_TENSOR(n) __builtin_amdgcn_s_wait_tensorcnt(n)
// 2D bf16 tile DMA: global (row-major, stride_elems) -> LDS rows padded by
// pad_amount+1 dwords every 2^(pad_interval+1) dwords. ISA 8.3/8.4 D# layout.
static __device__ __forceinline__ void tdm_load_2d(
    unsigned lds_addr, const __bf16* g, unsigned tile_w, unsigned tile_h,
    unsigned stride_elems, unsigned pad_interval, unsigned pad_amount) {
  unsigned long long ga = (unsigned long long)(size_t)g;
  v4u_vec g0 = {0u, 0u, 0u, 0u};
  g0[0] = 1u;                                   // count=1, user descriptor
  g0[1] = lds_addr;                             // LDS byte address
  g0[2] = (unsigned)ga;                         // global_addr[31:0]
  g0[3] = ((unsigned)(ga >> 32) & 0x01FFFFFFu)  // global_addr[56:32]
          | (2u << 30);                         // type=2 ("image")
  v8i_vec g1 = {0, 0, 0, 0, 0, 0, 0, 0};
  g1[0] = (int)((1u << 16)                      // data_size=1 -> 2 bytes
                | (1u << 20)                    // pad_enable
                | (pad_interval << 22) | (pad_amount << 25));
  g1[1] = (int)((tile_w & 0xFFFFu) << 16);      // tensor_dim0[15:0] @bits63:48
  g1[2] = (int)(((tile_w >> 16) & 0xFFFFu) | ((tile_h & 0xFFFFu) << 16)); // td0 hi | td1 lo
  g1[3] = (int)(((tile_h >> 16) & 0xFFFFu) | ((tile_w & 0xFFFFu) << 16)); // td1 hi | tile_dim0
  g1[4] = (int)(tile_h & 0xFFFFu);              // tile_dim1 (tile_dim2=0)
  g1[5] = (int)stride_elems;                    // tensor_dim0_stride[31:0]
  v4i_vec g2 = {0, 0, 0, 0};
  v4i_vec g3 = {0, 0, 0, 0};
  v8i_vec g4 = {0, 0, 0, 0, 0, 0, 0, 0};
  __builtin_amdgcn_tensor_load_to_lds(g0, g1, g2, g3, g4, 0);
}
#endif

static __device__ __forceinline__ v16bf combine16(v8bf lo, v8bf hi) {
  union { v16bf v; v8bf h[2]; } u;
  u.h[0] = lo; u.h[1] = hi;
  return u.v;
}

// ---------------------------------------------------------------------------
// LayerNorm (ddof=1). One block per row, D=768. fp32 + bf16 outputs.
// ---------------------------------------------------------------------------
__global__ __launch_bounds__(256) void ln_kernel(
    const float* __restrict__ x, const float* __restrict__ g,
    const float* __restrict__ b, float* __restrict__ yf,
    __bf16* __restrict__ yb, int D) {
  const int row = blockIdx.x;
  const float* xr = x + (size_t)row * D;
  __shared__ float red[256];

  float s = 0.f;
  for (int i = threadIdx.x; i < D; i += 256) s += xr[i];
  red[threadIdx.x] = s;
  __syncthreads();
  for (int off = 128; off > 0; off >>= 1) {
    if (threadIdx.x < off) red[threadIdx.x] += red[threadIdx.x + off];
    __syncthreads();
  }
  const float mean = red[0] / (float)D;
  __syncthreads();

  float s2 = 0.f;
  for (int i = threadIdx.x; i < D; i += 256) {
    float d = xr[i] - mean;
    s2 += d * d;
  }
  red[threadIdx.x] = s2;
  __syncthreads();
  for (int off = 128; off > 0; off >>= 1) {
    if (threadIdx.x < off) red[threadIdx.x] += red[threadIdx.x + off];
    __syncthreads();
  }
  const float var = red[0] / (float)(D - 1);
  const float inv = rsqrtf(var + 1e-5f);
  for (int i = threadIdx.x; i < D; i += 256) {
    float v = g[i] * (xr[i] - mean) * inv + b[i];
    yf[(size_t)row * D + i] = v;
    yb[(size_t)row * D + i] = (__bf16)v;
  }
}

// ---------------------------------------------------------------------------
// Weight packing (transposed [N][K] bf16 -> contiguous WMMA B fragments).
// Q columns pre-scaled by 1/sqrt(hd)=0.125 so attention needs no score scale.
// ---------------------------------------------------------------------------
__global__ void pack_qkv_w(const float* __restrict__ Wq, const float* __restrict__ Wk,
                           const float* __restrict__ Wv, __bf16* __restrict__ out) {
  int idx = blockIdx.x * 256 + threadIdx.x;
  if (idx >= 2304 * 768) return;
  int n = idx / 768, d = idx % 768;
  const float* W;
  int nn = n;
  float scl = 1.f;
  if (n < 768)       { W = Wq; scl = 0.125f; }
  else if (n < 1536) { W = Wk; nn = n - 768; }
  else               { W = Wv; nn = n - 1536; }
  int h = nn >> 6, kk = nn & 63;
  out[idx] = (__bf16)(W[((size_t)h * 768 + d) * 64 + kk] * scl);
}

__global__ void pack_qkv_b(const float* __restrict__ bq, const float* __restrict__ bk,
                           const float* __restrict__ bv, float* __restrict__ out) {
  int i = blockIdx.x * 256 + threadIdx.x;
  if (i < 768)       out[i] = bq[i] * 0.125f;
  else if (i < 1536) out[i] = bk[i - 768];
  else if (i < 2304) out[i] = bv[i - 1536];
}

__global__ void pack_transpose(const float* __restrict__ src, __bf16* __restrict__ dst,
                               int R, int C) {
  int idx = blockIdx.x * 256 + threadIdx.x;
  if (idx >= R * C) return;
  int c = idx / R, r = idx % R;
  dst[idx] = (__bf16)src[(size_t)r * C + c];
}

// ---------------------------------------------------------------------------
// Tiled bf16 GEMM: out = [res +] [gelu](A[M,K] @ Bt[N,K]^T + bias[N])
// 128x128 tile, BK=32, double-buffered. A tile: per-lane async->LDS.
// B tile: one TDM descriptor issued by wave 0 (pad writes the 40-half rows).
// QKVSPLIT: cols<1536 -> qk[8192,1536]; cols>=1536 -> vT[B, H*hd, S].
// ---------------------------------------------------------------------------
template <bool GELU, bool HASRES, bool DOF32, bool DOBF16, bool QKVSPLIT>
__global__ __launch_bounds__(256) void gemm_bf16(
    const __bf16* __restrict__ A, const __bf16* __restrict__ Bt,
    const float* __restrict__ bias, const float* __restrict__ res,
    float* __restrict__ outF, __bf16* __restrict__ outB,
    __bf16* __restrict__ vtB, int M, int N, int K) {
  __shared__ __align__(16) __bf16 sA[2][128][40];
  __shared__ __align__(16) __bf16 sB[2][128][40];

  const int tid  = threadIdx.x;
  const int lane = tid & 31, w = tid >> 5;
  const int wm = w >> 1, wn = w & 1;
  const int m0 = blockIdx.y * 128, n0 = blockIdx.x * 128;
  const int lh = lane >> 4, ll = lane & 15;
  const int r = tid >> 1, seg = (tid & 1) * 16;

  v8f zero = {0.f, 0.f, 0.f, 0.f, 0.f, 0.f, 0.f, 0.f};
  v8f acc[2][4];
  for (int i = 0; i < 2; i++)
    for (int j = 0; j < 4; j++) acc[i][j] = zero;

  auto stageA = [&](int buf, int k0) {
    const __bf16* ga = A + (size_t)(m0 + r) * K + k0 + seg;
#ifdef HAVE_ASYNC_LDS
    async_cp128(ga,     &sA[buf][r][seg]);
    async_cp128(ga + 8, &sA[buf][r][seg + 8]);
#else
    v8bf a0 = *(const v8bf*)ga, a1 = *(const v8bf*)(ga + 8);
    *(v8bf*)&sA[buf][r][seg] = a0;  *(v8bf*)&sA[buf][r][seg + 8] = a1;
#endif
  };
  auto stageB = [&](int buf, int k0) {
#ifdef HAVE_TDM
    if (w == 0)  // rows of 32 elems = 16 dwords (interval 3), pad 4 dwords (3)
      tdm_load_2d(lds_addr_of(&sB[buf][0][0]), Bt + (size_t)n0 * K + k0,
                  32, 128, (unsigned)K, 3, 3);
#else
    const __bf16* gb = Bt + (size_t)(n0 + r) * K + k0 + seg;
#ifdef HAVE_ASYNC_LDS
    async_cp128(gb,     &sB[buf][r][seg]);
    async_cp128(gb + 8, &sB[buf][r][seg + 8]);
#else
    v8bf b0 = *(const v8bf*)gb, b1 = *(const v8bf*)(gb + 8);
    *(v8bf*)&sB[buf][r][seg] = b0;  *(v8bf*)&sB[buf][r][seg + 8] = b1;
#endif
#endif
  };

#ifdef HAVE_TDM
#define GEMM_PREF_OPS 2
#else
#define GEMM_PREF_OPS 4
#endif

  const int nk = K >> 5;
  stageA(0, 0);
  stageB(0, 0);
  for (int i = 0; i < nk; i++) {
    const int buf = i & 1;
    if (i + 1 < nk) {
      stageA(buf ^ 1, (i + 1) << 5);
      stageB(buf ^ 1, (i + 1) << 5);
#ifdef HAVE_ASYNC_LDS
      WAIT_ASYNC(GEMM_PREF_OPS);
#endif
#ifdef HAVE_TDM
      if (w == 0) WAIT_TENSOR(1);
#endif
    } else {
#ifdef HAVE_ASYNC_LDS
      WAIT_ASYNC(0);
#endif
#ifdef HAVE_TDM
      if (w == 0) WAIT_TENSOR(0);
#endif
    }
    __syncthreads();

    v16bf af[2], bfrag[4];
    for (int ms = 0; ms < 2; ms++) {
      int m = wm * 32 + ms * 16 + ll;
      int ko = lh * 8;
      af[ms] = combine16(*(const v8bf*)&sA[buf][m][ko],
                         *(const v8bf*)&sA[buf][m][16 + ko]);
    }
    for (int ns = 0; ns < 4; ns++) {
      int n = wn * 64 + ns * 16 + ll;
      int ko = lh * 8;
      bfrag[ns] = combine16(*(const v8bf*)&sB[buf][n][ko],
                            *(const v8bf*)&sB[buf][n][ko + 16]);
    }
    for (int ms = 0; ms < 2; ms++)
      for (int ns = 0; ns < 4; ns++)
        acc[ms][ns] = WMMA_BF16(af[ms], bfrag[ns], acc[ms][ns]);
    __syncthreads();
  }

  for (int ms = 0; ms < 2; ms++) {
    for (int ns = 0; ns < 4; ns++) {
      const int col = n0 + wn * 64 + ns * 16 + ll;
      const float bv = bias[col];
      const int row0 = m0 + wm * 32 + ms * 16 + lh * 8;
      for (int v = 0; v < 8; v++) {
        const int row = row0 + v;
        float val = acc[ms][ns][v] + bv;
        if constexpr (GELU)   val = 0.5f * val * (1.f + erff(val * 0.70710678118654752f));
        if constexpr (HASRES) val += res[(size_t)row * N + col];
        if constexpr (DOF32)  outF[(size_t)row * N + col] = val;
        if constexpr (DOBF16) outB[(size_t)row * N + col] = (__bf16)val;
        if constexpr (QKVSPLIT) {
          if (n0 < 1536) {  // q|k -> [8192,1536]
            outB[(size_t)row * 1536 + col] = (__bf16)val;
          } else {          // v -> transposed [B][H*hd][S]
            vtB[((size_t)(row >> 10) * 768 + (col - 1536)) * 1024 + (row & 1023)] =
                (__bf16)val;
          }
        }
      }
    }
  }
}

// ---------------------------------------------------------------------------
// Flash attention. qk = [B*S,1536] bf16 (q pre-scaled | k), vt = V^T bf16.
// Block = 128 q rows of one (b,h); wave = 16 q rows x full 64-wide KV block.
// K/V^T tiles double-buffered, staged by TDM (wave 0) into padded LDS.
// ---------------------------------------------------------------------------
__global__ __launch_bounds__(256) void attn_kernel(
    const __bf16* __restrict__ qk, const __bf16* __restrict__ vt,
    __bf16* __restrict__ o, int S) {
  const int h = blockIdx.y, bb = blockIdx.z;
  const int q0 = blockIdx.x * 128;
  const int tid = threadIdx.x, lane = tid & 31, w = tid >> 5;
  const int lh = lane >> 4, ll = lane & 15;
  const int rowbase = bb * S;

  __shared__ __align__(16) __bf16 sK[2][64][72];
  __shared__ __align__(16) __bf16 sVt[2][64][72];
  __shared__ __align__(16) __bf16 sP[8][16][72];

  v16bf qf[2];
  {
    int m = q0 + w * 16 + ll;
    const __bf16* qrow = qk + (size_t)(rowbase + m) * 1536 + h * 64;
    for (int ks = 0; ks < 2; ks++) {
      int ko = ks * 32 + lh * 8;
      qf[ks] = combine16(*(const v8bf*)(qrow + ko), *(const v8bf*)(qrow + ko + 16));
    }
  }

  v8f zero = {0.f, 0.f, 0.f, 0.f, 0.f, 0.f, 0.f, 0.f};
  v8f o_acc[4];
  for (int i = 0; i < 4; i++) o_acc[i] = zero;
  float mrow[8], lrow[8];
  for (int v = 0; v < 8; v++) { mrow[v] = -1e30f; lrow[v] = 0.f; }

  const int lr = tid >> 2, lpart = (tid & 3) * 16;
  const __bf16* vhead = vt + ((size_t)bb * 768 + h * 64) * 1024;

  auto stageKV = [&](int buf, int kv0) {
#ifdef HAVE_TDM
    if (w == 0) {  // rows of 64 elems = 32 dwords (interval 4), pad 4 dwords
      tdm_load_2d(lds_addr_of(&sK[buf][0][0]),
                  qk + (size_t)(rowbase + kv0) * 1536 + 768 + h * 64,
                  64, 64, 1536, 4, 3);
      tdm_load_2d(lds_addr_of(&sVt[buf][0][0]), vhead + kv0, 64, 64, 1024, 4, 3);
    }
#else
    const __bf16* krow = qk + (size_t)(rowbase + kv0 + lr) * 1536 + 768 + h * 64 + lpart;
    const __bf16* vrow = vhead + (size_t)lr * 1024 + kv0 + lpart;
#ifdef HAVE_ASYNC_LDS
    async_cp128(krow,     &sK[buf][lr][lpart]);
    async_cp128(krow + 8, &sK[buf][lr][lpart + 8]);
    async_cp128(vrow,     &sVt[buf][lr][lpart]);
    async_cp128(vrow + 8, &sVt[buf][lr][lpart + 8]);
#else
    *(v8bf*)&sK[buf][lr][lpart]      = *(const v8bf*)krow;
    *(v8bf*)&sK[buf][lr][lpart + 8]  = *(const v8bf*)(krow + 8);
    *(v8bf*)&sVt[buf][lr][lpart]     = *(const v8bf*)vrow;
    *(v8bf*)&sVt[buf][lr][lpart + 8] = *(const v8bf*)(vrow + 8);
#endif
#endif
  };

  const int nkv = S >> 6;
  stageKV(0, 0);
  for (int it = 0; it < nkv; it++) {
    const int buf = it & 1;
    if (it + 1 < nkv) {
      stageKV(buf ^ 1, (it + 1) << 6);
#ifdef HAVE_TDM
      if (w == 0) WAIT_TENSOR(2);
#elif defined(HAVE_ASYNC_LDS)
      WAIT_ASYNC(4);
#endif
    } else {
#ifdef HAVE_TDM
      if (w == 0) WAIT_TENSOR(0);
#elif defined(HAVE_ASYNC_LDS)
      WAIT_ASYNC(0);
#endif
    }
    __syncthreads();

    // scores (Q already carries the 1/sqrt(hd) scale)
    v8f sc[4];
    for (int nt = 0; nt < 4; nt++) {
      v8f c = zero;
      for (int ks = 0; ks < 2; ks++) {
        int kvr = nt * 16 + ll;
        int ko = ks * 32 + lh * 8;
        v16bf kf = combine16(*(const v8bf*)&sK[buf][kvr][ko],
                             *(const v8bf*)&sK[buf][kvr][ko + 16]);
        c = WMMA_BF16(qf[ks], kf, c);
      }
      sc[nt] = c;
    }

    float pexp[4][8];
    for (int v = 0; v < 8; v++) {
      float pm = fmaxf(fmaxf(sc[0][v], sc[1][v]), fmaxf(sc[2][v], sc[3][v]));
      pm = fmaxf(pm, __shfl_xor(pm, 1, 32));
      pm = fmaxf(pm, __shfl_xor(pm, 2, 32));
      pm = fmaxf(pm, __shfl_xor(pm, 4, 32));
      pm = fmaxf(pm, __shfl_xor(pm, 8, 32));
      float mn = fmaxf(mrow[v], pm);
      float scale = __expf(mrow[v] - mn);
      float rs = 0.f;
      for (int nt = 0; nt < 4; nt++) {
        float e = __expf(sc[nt][v] - mn);
        pexp[nt][v] = e;
        rs += e;
      }
      rs += __shfl_xor(rs, 1, 32);
      rs += __shfl_xor(rs, 2, 32);
      rs += __shfl_xor(rs, 4, 32);
      rs += __shfl_xor(rs, 8, 32);
      lrow[v] = lrow[v] * scale + rs;
      mrow[v] = mn;
      for (int nt = 0; nt < 4; nt++) o_acc[nt][v] *= scale;
    }

    for (int nt = 0; nt < 4; nt++)
      for (int v = 0; v < 8; v++)
        sP[w][lh * 8 + v][nt * 16 + ll] = (__bf16)pexp[nt][v];
    v16bf pf[2];
    for (int ks = 0; ks < 2; ks++) {
      int ko = ks * 32 + lh * 8;
      pf[ks] = combine16(*(const v8bf*)&sP[w][ll][ko], *(const v8bf*)&sP[w][ll][ko + 16]);
    }

    for (int nt = 0; nt < 4; nt++) {
      for (int ks = 0; ks < 2; ks++) {
        int hc = nt * 16 + ll;
        int ko = ks * 32 + lh * 8;
        v16bf vf = combine16(*(const v8bf*)&sVt[buf][hc][ko],
                             *(const v8bf*)&sVt[buf][hc][ko + 16]);
        o_acc[nt] = WMMA_BF16(pf[ks], vf, o_acc[nt]);
      }
    }
    __syncthreads();  // protect buf before next prefetch overwrites it
  }

  for (int v = 0; v < 8; v++) {
    float inv = 1.f / lrow[v];
    int row = rowbase + q0 + w * 16 + lh * 8 + v;
    for (int nt = 0; nt < 4; nt++) {
      int col = h * 64 + nt * 16 + ll;
      o[(size_t)row * 768 + col] = (__bf16)(o_acc[nt][v] * inv);
    }
  }
}

// ---------------------------------------------------------------------------
// Orchestration. B=8, S=1024, D=768, H=12, hd=64, FF=3072, M = 8192.
// ---------------------------------------------------------------------------
extern "C" void kernel_launch(void* const* d_in, const int* in_sizes, int n_in,
                              void* d_out, int out_size, void* d_ws, size_t ws_size,
                              hipStream_t stream) {
  (void)in_sizes; (void)n_in; (void)out_size; (void)ws_size;
  const float* x   = (const float*)d_in[0];
  const float* Wq  = (const float*)d_in[1];
  const float* bq  = (const float*)d_in[2];
  const float* Wk  = (const float*)d_in[3];
  const float* bk  = (const float*)d_in[4];
  const float* Wv  = (const float*)d_in[5];
  const float* bv  = (const float*)d_in[6];
  const float* Wo  = (const float*)d_in[7];
  const float* bo  = (const float*)d_in[8];
  const float* W1  = (const float*)d_in[9];
  const float* b1  = (const float*)d_in[10];
  const float* W2  = (const float*)d_in[11];
  const float* b2  = (const float*)d_in[12];
  const float* g1  = (const float*)d_in[13];
  const float* be1 = (const float*)d_in[14];
  const float* g2  = (const float*)d_in[15];
  const float* be2 = (const float*)d_in[16];

  char* ws = (char*)d_ws;
  float*  xln_f  = (float*) (ws + 0);          // 25165824
  __bf16* xln_b  = (__bf16*)(ws + 25165824);   // 12582912
  float*  x2_f   = (float*) (ws + 37748736);   // 25165824
  __bf16* wqkv_t = (__bf16*)(ws + 62914560);   // 3538944
  __bf16* wo_t   = (__bf16*)(ws + 66453504);   // 1179648
  __bf16* w1_t   = (__bf16*)(ws + 67633152);   // 4718592
  __bf16* w2_t   = (__bf16*)(ws + 72351744);   // 4718592
  float*  bqkv   = (float*) (ws + 77070336);   // 9216
  __bf16* qk     = (__bf16*)(ws + 77079552);   // 8192*1536*2 = 25165824
  __bf16* vt     = (__bf16*)(ws + 102245376);  // 8*768*1024*2 = 12582912
  __bf16* attn_o = (__bf16*)(ws + 114828288);  // 12582912
  __bf16* h_b    = (__bf16*)(ws + 77079552);   // reuse qk+vt+attn_o region

  pack_qkv_w<<<(2304 * 768 + 255) / 256, 256, 0, stream>>>(Wq, Wk, Wv, wqkv_t);
  pack_qkv_b<<<(2304 + 255) / 256, 256, 0, stream>>>(bq, bk, bv, bqkv);
  pack_transpose<<<(768 * 768 + 255) / 256, 256, 0, stream>>>(Wo, wo_t, 768, 768);
  pack_transpose<<<(768 * 3072 + 255) / 256, 256, 0, stream>>>(W1, w1_t, 768, 3072);
  pack_transpose<<<(3072 * 768 + 255) / 256, 256, 0, stream>>>(W2, w2_t, 3072, 768);

  // 1) x = ln1(x)
  ln_kernel<<<8192, 256, 0, stream>>>(x, g1, be1, xln_f, xln_b, 768);
  // 2) fused QKV projection -> qk (bf16, q pre-scaled) + vT (bf16)
  gemm_bf16<false, false, false, false, true><<<dim3(18, 64), 256, 0, stream>>>(
      xln_b, wqkv_t, bqkv, nullptr, nullptr, qk, vt, 8192, 2304, 768);
  // 3) attention
  attn_kernel<<<dim3(8, 12, 8), 256, 0, stream>>>(qk, vt, attn_o, 1024);
  // 4) x = xln1 + attn @ Wo + bo
  gemm_bf16<false, true, true, false, false><<<dim3(6, 64), 256, 0, stream>>>(
      attn_o, wo_t, bo, xln_f, x2_f, nullptr, nullptr, 8192, 768, 768);
  // 5) x = ln2(x)
  ln_kernel<<<8192, 256, 0, stream>>>(x2_f, g2, be2, xln_f, xln_b, 768);
  // 6) h = gelu(x @ W1 + b1)
  gemm_bf16<true, false, false, true, false><<<dim3(24, 64), 256, 0, stream>>>(
      xln_b, w1_t, b1, nullptr, nullptr, h_b, nullptr, 8192, 3072, 768);
  // 7) out = x + h @ W2 + b2
  gemm_bf16<false, true, true, false, false><<<dim3(6, 64), 256, 0, stream>>>(
      h_b, w2_t, b2, xln_f, (float*)d_out, nullptr, nullptr, 8192, 768, 3072);
}
